// ModelMamba_11072425689899
// MI455X (gfx1250) — matrix operand order repaired and checked by reading, not compiled, not run on hardware
//
#include <hip/hip_runtime.h>

// ---------------------------------------------------------------------------
// Model dims (compile-time constants from the reference)
// ---------------------------------------------------------------------------
#define BB 8
#define LL 2048
#define DMODEL 512
#define DINNER 1024
#define DSTATE 16
#define DCONV 4
#define NLAYER 4
#define DTRANK 32
#define DHID 256
#define MROWS (BB * LL)   // 16384

// WMMA types (probe-confirmed signatures for gfx1250)
typedef __attribute__((ext_vector_type(16))) __bf16 v16bf;
typedef __attribute__((ext_vector_type(8)))  float  v8f;

__device__ __forceinline__ unsigned short f2bf(float f) {
    unsigned int u = __float_as_uint(f);
    u += 0x7FFFu + ((u >> 16) & 1u);   // round to nearest even
    return (unsigned short)(u >> 16);
}

// gfx1250 async copy: global memory -> LDS, 16B per lane, tracked by ASYNCcnt
__device__ __forceinline__ void async_b128(unsigned lds_off, const void* gaddr) {
    asm volatile("global_load_async_to_lds_b128 %0, %1, off"
                 :: "v"(lds_off), "v"(gaddr)
                 : "memory");
}

// ---------------------------------------------------------------------------
// f32 -> bf16 bulk convert (weights)
// ---------------------------------------------------------------------------
__global__ __launch_bounds__(256) void f32_to_bf16_kernel(
    const float* __restrict__ in, unsigned short* __restrict__ out, size_t n) {
    size_t i = (size_t)blockIdx.x * 256 + threadIdx.x;
    if (i < n) out[i] = f2bf(in[i]);
}

// ---------------------------------------------------------------------------
// Embedding: x = mask * (seq_emb[rna] + tissue_emb[tissue]); writes f32 + bf16
// ---------------------------------------------------------------------------
__global__ __launch_bounds__(256) void embed_kernel(
    const int* __restrict__ rna, const int* __restrict__ tissue,
    const float* __restrict__ seq_emb, const float* __restrict__ tis_emb,
    float* __restrict__ x, unsigned short* __restrict__ xb) {
    int flat = blockIdx.x * 256 + threadIdx.x;      // B*L*DMODEL = 8388608
    int d = flat & (DMODEL - 1);
    int l = (flat >> 9) & (LL - 1);
    int b = flat >> 20;
    int tok = rna[b * LL + l];
    float v = 0.f;
    if (tok != 0)
        v = seq_emb[tok * DMODEL + d] + tis_emb[tissue[b] * DMODEL + d];
    x[flat]  = v;
    xb[flat] = f2bf(v);
}

// ---------------------------------------------------------------------------
// Generic bf16 WMMA GEMM:  C[M,N] = A[M,K] * W[N,K]^T
//   block tile 128x64, 8 waves (wave32), K step 32
//   double-buffered LDS filled with GLOBAL_LOAD_ASYNC_TO_LDS_B128 (ASYNCcnt),
//   v_wmma_f32_16x16x32_bf16 accumulate.
//   mode 0: store f32        mode 1: softplus(x + bias[n]) f32
//   mode 2: store f32 and bf16 (fused conversion for next GEMM)
// ---------------------------------------------------------------------------
#define TM 128
#define TN 64
#define TK 32

__global__ __launch_bounds__(256) void gemm_bf16_wmma(
    const unsigned short* __restrict__ A, int lda,
    const unsigned short* __restrict__ W,          // [N,K] row-major (B^T)
    const float* __restrict__ bias,
    float* __restrict__ C, unsigned short* __restrict__ Cb,
    int M, int N, int K, int mode) {
    __shared__ __align__(16) unsigned short sA[2][TM * TK];   // 2 x 8 KB
    __shared__ __align__(16) unsigned short sB[2][TN * TK];   // 2 x 4 KB

    const int tid  = threadIdx.x;
    const int lane = tid & 31;
    const int wave = tid >> 5;
    const int m0 = blockIdx.y * TM;
    const int n0 = blockIdx.x * TN;

    v8f zero = {0.f, 0.f, 0.f, 0.f, 0.f, 0.f, 0.f, 0.f};
    v8f acc[4];
#pragma unroll
    for (int t = 0; t < 4; ++t) acc[t] = zero;

    // cooperative tile-fill mapping (16B async chunks per thread)
    const int arow  = tid >> 1;          // 0..127
    const int ahalf = (tid & 1) * 16;    // shorts: 0 or 16
    const int brow  = tid >> 2;          // 0..63
    const int bq    = (tid & 3) * 8;     // shorts: 0,8,16,24

    const unsigned short* gA = A + (size_t)(m0 + arow) * lda + ahalf;
    const unsigned short* gB = W + (size_t)(n0 + brow) * K + bq;

    const unsigned ldsA0 = (unsigned)(arow * TK + ahalf) * 2u;
    const unsigned ldsB0 = (unsigned)(brow * TK + bq) * 2u;
    const unsigned baseA[2] = {(unsigned)(size_t)&sA[0][0], (unsigned)(size_t)&sA[1][0]};
    const unsigned baseB[2] = {(unsigned)(size_t)&sB[0][0], (unsigned)(size_t)&sB[1][0]};

    auto issue_tile = [&](int kb, int buf) {
        async_b128(baseA[buf] + ldsA0,      gA + kb);
        async_b128(baseA[buf] + ldsA0 + 16, gA + kb + 8);
        async_b128(baseB[buf] + ldsB0,      gB + kb);
    };

    const int nk = K / TK;
    issue_tile(0, 0);
    if (nk > 1) issue_tile(TK, 1);

    const int kh = lane >> 4;            // K-half select (WMMA lane layout)
    const int ln = lane & 15;

    for (int i = 0; i < nk; ++i) {
        const int buf = i & 1;
        if (i + 1 < nk) {
            asm volatile("s_wait_asynccnt 0x3" ::: "memory");  // tile i landed
        } else {
            asm volatile("s_wait_asynccnt 0x0" ::: "memory");
        }
        __syncthreads();

        // A fragment: 16x32 bf16; lane ln holds row M=ln, K pairs per ISA layout
        union { v16bf v; unsigned int u[8]; } af;
        const int rA = wave * 16 + ln;
#pragma unroll
        for (int j = 0; j < 8; ++j) {
            int k0 = (j < 4) ? (2 * j + 8 * kh) : (16 + 2 * (j - 4) + 8 * kh);
            af.u[j] = *(const unsigned int*)&sA[buf][rA * TK + k0];
        }
#pragma unroll
        for (int t = 0; t < 4; ++t) {
            // B fragment: 32x16 bf16; lane ln holds column N=ln, K pairs per ISA layout (same as A)
            union { v16bf v; unsigned int u[8]; } bf;
            const int rB = t * 16 + ln;
#pragma unroll
            for (int j = 0; j < 8; ++j) {
                int k0 = (j < 4) ? (2 * j + 8 * kh) : (16 + 2 * (j - 4) + 8 * kh);
                bf.u[j] = *(const unsigned int*)&sB[buf][rB * TK + k0];
            }
            acc[t] = __builtin_amdgcn_wmma_f32_16x16x32_bf16(
                false, af.v, false, bf.v, (short)0, acc[t], false, false);
        }
        __syncthreads();                 // all waves done reading buf
        if (i + 2 < nk) issue_tile((i + 2) * TK, buf);
    }

    // epilogue: C/D layout — lane ln = column, VGPR j = row j + 8*kh
#pragma unroll
    for (int t = 0; t < 4; ++t) {
        int col = n0 + t * 16 + ln;
#pragma unroll
        for (int j = 0; j < 8; ++j) {
            int row = m0 + wave * 16 + j + 8 * kh;
            float v = acc[t][j];
            if (mode == 1) {
                v += bias[col];
                v = (v > 20.f) ? v : log1pf(__expf(v));
            }
            size_t off = (size_t)row * N + col;
            C[off] = v;
            if (mode == 2) Cb[off] = f2bf(v);
        }
    }
}

// ---------------------------------------------------------------------------
// Causal depthwise conv (k=4) + bias + SiLU; writes u f32 + bf16
// ---------------------------------------------------------------------------
__global__ __launch_bounds__(256) void conv_silu_kernel(
    const float* __restrict__ xz, const float* __restrict__ cw,
    const float* __restrict__ cb,
    float* __restrict__ u, unsigned short* __restrict__ ub) {
    int flat = blockIdx.x * 256 + threadIdx.x;      // B*L*DINNER = 16777216
    int d = flat & (DINNER - 1);
    int l = (flat >> 10) & (LL - 1);
    int b = flat >> 21;
    float acc = cb[d];
#pragma unroll
    for (int j = 0; j < DCONV; ++j) {
        int t = l + j - (DCONV - 1);
        if (t >= 0)
            acc += cw[d * DCONV + j] * xz[((size_t)(b * LL + t)) * (2 * DINNER) + d];
    }
    float s = acc / (1.f + __expf(-acc));           // silu
    u[flat]  = s;
    ub[flat] = f2bf(s);
}

// ---------------------------------------------------------------------------
// Selective scan: one thread per (b,d); h[16] in registers.
// B_t/C_t are wave-uniform -> staged through LDS once per block per step.
// Fuses +u*D and *silu(z); writes y as bf16 (input to out-proj GEMM).
// ---------------------------------------------------------------------------
__global__ __launch_bounds__(256) void scan_kernel(
    const float* __restrict__ dt, const float* __restrict__ u,
    const float* __restrict__ xdbl, const float* __restrict__ xz,
    const float* __restrict__ Alog, const float* __restrict__ Dp,
    unsigned short* __restrict__ yb) {
    __shared__ float sBC[2 * DSTATE];
    int flat = blockIdx.x * 256 + threadIdx.x;      // B*DINNER = 8192
    int d = flat & (DINNER - 1);
    int b = flat >> 10;                             // all threads in block: same b
    float A[DSTATE], h[DSTATE];
#pragma unroll
    for (int n = 0; n < DSTATE; ++n) {
        A[n] = -__expf(Alog[d * DSTATE + n]);
        h[n] = 0.f;
    }
    const float Dd = Dp[d];
    for (int l = 0; l < LL; ++l) {
        size_t row = (size_t)(b * LL + l);
        if (threadIdx.x < 2 * DSTATE)
            sBC[threadIdx.x] = xdbl[row * 64 + DTRANK + threadIdx.x];
        float dtv = dt[row * DINNER + d];
        float uv  = u[row * DINNER + d];
        float z   = xz[row * (2 * DINNER) + DINNER + d];
        __syncthreads();
        float du = dtv * uv;
        float y = 0.f;
#pragma unroll
        for (int n = 0; n < DSTATE; ++n) {
            float dA = __expf(dtv * A[n]);
            h[n] = dA * h[n] + du * sBC[n];
            y += h[n] * sBC[DSTATE + n];
        }
        __syncthreads();
        y += uv * Dd;
        y *= z / (1.f + __expf(-z));                // * silu(z)
        yb[row * DINNER + d] = f2bf(y);
    }
}

// ---------------------------------------------------------------------------
// Head: gather last token, 512->256 relu MLP, 256->1
// ---------------------------------------------------------------------------
__global__ __launch_bounds__(256) void head_kernel(
    const float* __restrict__ x, const int* __restrict__ lens,
    const float* __restrict__ W1, const float* __restrict__ b1,
    const float* __restrict__ W2, const float* __restrict__ b2,
    float* __restrict__ out) {
    __shared__ float red[DHID];
    int b = blockIdx.x;
    int j = threadIdx.x;
    int last = lens[b] - 1;
    const float* xl = x + ((size_t)(b * LL + last)) * DMODEL;
    float acc = b1[j];
    for (int d = 0; d < DMODEL; ++d) acc += xl[d] * W1[j * DMODEL + d];
    acc = fmaxf(acc, 0.f);
    red[j] = acc * W2[j];
    __syncthreads();
    for (int s = DHID / 2; s > 0; s >>= 1) {
        if (j < s) red[j] += red[j + s];
        __syncthreads();
    }
    if (j == 0) out[b] = red[0] + b2[0];
}

// ---------------------------------------------------------------------------
// Launch
// ---------------------------------------------------------------------------
extern "C" void kernel_launch(void* const* d_in, const int* in_sizes, int n_in,
                              void* d_out, int out_size, void* d_ws, size_t ws_size,
                              hipStream_t stream) {
    const int*   rna     = (const int*)d_in[0];
    const int*   tissue  = (const int*)d_in[1];
    const int*   lens    = (const int*)d_in[2];
    const float* seq_emb = (const float*)d_in[3];
    const float* tis_emb = (const float*)d_in[4];
    const float* W_in    = (const float*)d_in[5];
    const float* conv_w  = (const float*)d_in[6];
    const float* conv_b  = (const float*)d_in[7];
    const float* W_xproj = (const float*)d_in[8];
    const float* W_dt    = (const float*)d_in[9];
    const float* b_dt    = (const float*)d_in[10];
    const float* A_log   = (const float*)d_in[11];
    const float* D_par   = (const float*)d_in[12];
    const float* W_out   = (const float*)d_in[13];
    const float* W1      = (const float*)d_in[14];
    const float* b1      = (const float*)d_in[15];
    const float* W2      = (const float*)d_in[16];
    const float* b2      = (const float*)d_in[17];
    float* out = (float*)d_out;

    size_t off = 0;
    auto alloc = [&](size_t bytes) -> void* {
        void* p = (void*)((char*)d_ws + off);
        off += (bytes + 255) & ~(size_t)255;
        return p;
    };
    float*          x     = (float*)alloc((size_t)MROWS * DMODEL * 4);
    unsigned short* xb    = (unsigned short*)alloc((size_t)MROWS * DMODEL * 2);
    float*          xz    = (float*)alloc((size_t)MROWS * 2 * DINNER * 4);
    float*          u     = (float*)alloc((size_t)MROWS * DINNER * 4);
    unsigned short* ub    = (unsigned short*)alloc((size_t)MROWS * DINNER * 2); // also y_bf16
    float*          xdbl  = (float*)alloc((size_t)MROWS * 64 * 4);
    unsigned short* xdblb = (unsigned short*)alloc((size_t)MROWS * 64 * 2);
    float*          dtb   = (float*)alloc((size_t)MROWS * DINNER * 4);
    unsigned short* Wib   = (unsigned short*)alloc((size_t)NLAYER * 2 * DINNER * DMODEL * 2);
    unsigned short* Wxb   = (unsigned short*)alloc((size_t)NLAYER * 64 * DINNER * 2);
    unsigned short* Wdtb  = (unsigned short*)alloc((size_t)NLAYER * DINNER * DTRANK * 2);
    unsigned short* Wob   = (unsigned short*)alloc((size_t)NLAYER * DMODEL * DINNER * 2);

    auto cvt = [&](const float* src, unsigned short* dst, size_t n) {
        f32_to_bf16_kernel<<<(unsigned)((n + 255) / 256), 256, 0, stream>>>(src, dst, n);
    };
    cvt(W_in,    Wib,  (size_t)NLAYER * 2 * DINNER * DMODEL);
    cvt(W_xproj, Wxb,  (size_t)NLAYER * 64 * DINNER);
    cvt(W_dt,    Wdtb, (size_t)NLAYER * DINNER * DTRANK);
    cvt(W_out,   Wob,  (size_t)NLAYER * DMODEL * DINNER);

    embed_kernel<<<(MROWS * DMODEL) / 256, 256, 0, stream>>>(
        rna, tissue, seq_emb, tis_emb, x, xb);

    for (int layer = 0; layer < NLAYER; ++layer) {
        // in-proj: xz[M, 2048] = x[M,512] * W_in^T
        gemm_bf16_wmma<<<dim3((2 * DINNER) / TN, MROWS / TM), 256, 0, stream>>>(
            xb, DMODEL, Wib + (size_t)layer * 2 * DINNER * DMODEL, nullptr,
            xz, nullptr, MROWS, 2 * DINNER, DMODEL, 0);
        // depthwise causal conv + silu
        conv_silu_kernel<<<(MROWS * DINNER) / 256, 256, 0, stream>>>(
            xz, conv_w + layer * DINNER * DCONV, conv_b + layer * DINNER, u, ub);
        // x-proj: xdbl[M,64] = u[M,1024] * W_xproj^T (f32 + bf16 out)
        gemm_bf16_wmma<<<dim3(64 / TN, MROWS / TM), 256, 0, stream>>>(
            ub, DINNER, Wxb + (size_t)layer * 64 * DINNER, nullptr,
            xdbl, xdblb, MROWS, 64, DINNER, 2);
        // dt: softplus(xdbl[:, :32] * W_dt^T + b_dt)
        gemm_bf16_wmma<<<dim3(DINNER / TN, MROWS / TM), 256, 0, stream>>>(
            xdblb, 64, Wdtb + (size_t)layer * DINNER * DTRANK, b_dt + layer * DINNER,
            dtb, nullptr, MROWS, DINNER, DTRANK, 1);
        // selective scan (+ u*D, * silu(z)); y -> bf16 into ub
        scan_kernel<<<(BB * DINNER) / 256, 256, 0, stream>>>(
            dtb, u, xdbl, xz, A_log + layer * DINNER * DSTATE, D_par + layer * DINNER, ub);
        // out-proj: x[M,512] = y[M,1024] * W_out^T (f32 + bf16 for next layer)
        gemm_bf16_wmma<<<dim3(DMODEL / TN, MROWS / TM), 256, 0, stream>>>(
            ub, DINNER, Wob + (size_t)layer * DMODEL * DINNER, nullptr,
            x, xb, MROWS, DMODEL, DINNER, 2);
    }

    head_kernel<<<BB, DHID, 0, stream>>>(x, lens, W1, b1, W2, b2, out);
}
